// SelfModifyingTitansLayer_37005438222350
// MI455X (gfx1250) — matrix operand order repaired and checked by reading, not compiled, not run on hardware
//
#include <hip/hip_runtime.h>
#include <math.h>

#define BT   32768   // B*T
#define HD   256     // H
#define SS   4096    // memory slots
#define EPSLN 1e-5f
#define WLR  0.1f

typedef __bf16 bf16;
typedef __attribute__((ext_vector_type(16))) __bf16 v16bf;
typedef __attribute__((ext_vector_type(8)))  __bf16 v8bf;
typedef __attribute__((ext_vector_type(8)))  float  v8f;
typedef int v4i_vs __attribute__((vector_size(16)));
typedef __attribute__((address_space(1))) v4i_vs* g_v4i_p;
typedef __attribute__((address_space(3))) v4i_vs* l_v4i_p;

__device__ __forceinline__ v8f wmma_bf16(v16bf a, v16bf b, v8f c) {
  return __builtin_amdgcn_wmma_f32_16x16x32_bf16(false, a, false, b, (short)0, c,
                                                 false, false);
}

// ---- async global -> LDS (CDNA5, ASYNCcnt-tracked) -------------------------
__device__ __forceinline__ void async_ld_b128(const void* g, void* l) {
#if __has_builtin(__builtin_amdgcn_global_load_async_to_lds_b128)
  __builtin_amdgcn_global_load_async_to_lds_b128(
      (g_v4i_p)(uintptr_t)g,
      (l_v4i_p)(uintptr_t)(uint32_t)(uintptr_t)l,  // LDS offset = addr[31:0]
      0, 0);
#else
  uint32_t lo = (uint32_t)(uintptr_t)l;
  asm volatile("global_load_async_to_lds_b128 %0, %1, off"
               :: "v"(lo), "v"(g) : "memory");
#endif
}
__device__ __forceinline__ void wait_async0() {
#if __has_builtin(__builtin_amdgcn_s_wait_asynccnt)
  __builtin_amdgcn_s_wait_asynccnt(0);
#else
  asm volatile("s_wait_asynccnt 0" ::: "memory");
#endif
}

// ---- fragment loaders (CDNA5 wave32 WMMA layouts, 16-bit data) -------------
// A: 16(M)x32(K), row-major, lane m=lane&15, half h=lane>>4
//    element j -> k = h*16+j
__device__ __forceinline__ v16bf load_a_frag(const bf16* A, int lda, int k0) {
  const int lane = threadIdx.x & 31;
  const int m = lane & 15, h = lane >> 4;
  const bf16* p = A + (size_t)m * lda + k0 + h * 16;
  v8bf lo = *(const v8bf*)p;
  v8bf hi = *(const v8bf*)(p + 8);
  return __builtin_shufflevector(lo, hi, 0,1,2,3,4,5,6,7,8,9,10,11,12,13,14,15);
}
// B: 32(K)x16(N) sourced from an N-major buffer Bt[n][k]; lane col n=lane&15,
//    element j -> k = (lane>>4)*16 + j  => one contiguous 32B read.
__device__ __forceinline__ v16bf load_b_frag(const bf16* Bt, int ldb, int n0, int k0) {
  const int lane = threadIdx.x & 31;
  const int n = n0 + (lane & 15), h = lane >> 4;
  return *(const v16bf*)(Bt + (size_t)n * ldb + k0 + h * 16);
}

__device__ __forceinline__ float gelu_exact(float x) {
  return 0.5f * x * (1.f + erff(x * 0.70710678118654752f));
}
__device__ __forceinline__ float redmax16(float v) {
#pragma unroll
  for (int off = 1; off < 16; off <<= 1) v = fmaxf(v, __shfl_xor(v, off, 16));
  return v;
}
__device__ __forceinline__ float redsum16(float v) {
#pragma unroll
  for (int off = 1; off < 16; off <<= 1) v += __shfl_xor(v, off, 16);
  return v;
}

// ---- small utility kernels -------------------------------------------------
__global__ void k_cvt(const float* __restrict__ in, bf16* __restrict__ out, int n) {
  int i = blockIdx.x * blockDim.x + threadIdx.x;
  if (i < n) out[i] = (bf16)in[i];
}
__global__ void k_trans(const float* __restrict__ W, bf16* __restrict__ Wt,
                        int R, int C) {  // W[R][C] -> Wt[C][R] (bf16)
  int i = blockIdx.x * blockDim.x + threadIdx.x;
  if (i >= R * C) return;
  int r = i / C, c = i - r * C;
  Wt[(size_t)c * R + r] = (bf16)W[i];
}
__global__ void k_zero(float* __restrict__ p, int n) {
  int i = blockIdx.x * blockDim.x + threadIdx.x;
  if (i < n) p[i] = 0.f;
}

// ---- GEMM1: x_proj = x @ W_in + b_in  (store f32 + bf16) -------------------
__global__ __launch_bounds__(256) void k_gemm1(
    const bf16* __restrict__ xb, const bf16* __restrict__ Wt,
    const float* __restrict__ bias, float* __restrict__ xproj,
    bf16* __restrict__ xpb) {
  const int mt = blockIdx.x, wave = threadIdx.x >> 5, lane = threadIdx.x & 31;
  const int n = lane & 15, h = lane >> 4, n0 = wave * 32;
  const bf16* A = xb + (size_t)mt * 16 * HD;
  v8f a0 = {}, a1 = {};
#pragma unroll
  for (int k0 = 0; k0 < HD; k0 += 32) {
    v16bf a = load_a_frag(A, HD, k0);
    a0 = wmma_bf16(a, load_b_frag(Wt, HD, n0, k0), a0);
    a1 = wmma_bf16(a, load_b_frag(Wt, HD, n0 + 16, k0), a1);
  }
#pragma unroll
  for (int r = 0; r < 8; ++r) {
    size_t row = (size_t)(mt * 16 + r + 8 * h) * HD;
    float v0 = a0[r] + bias[n0 + n];
    float v1 = a1[r] + bias[n0 + 16 + n];
    xproj[row + n0 + n] = v0;        xproj[row + n0 + 16 + n] = v1;
    xpb[row + n0 + n] = (bf16)v0;    xpb[row + n0 + 16 + n] = (bf16)v1;
  }
}

// ---- GEMM2: query = gelu(LN(x_proj @ W_q + b_q)) -> bf16 -------------------
__global__ __launch_bounds__(256) void k_gemm_q(
    const bf16* __restrict__ xpb, const bf16* __restrict__ Wqt,
    const float* __restrict__ bq, const float* __restrict__ gq,
    const float* __restrict__ beq, bf16* __restrict__ queryb) {
  __shared__ float red1[8][16], red2[8][16], smu[16], srs[16];
  const int mt = blockIdx.x, wave = threadIdx.x >> 5, lane = threadIdx.x & 31;
  const int n = lane & 15, h = lane >> 4, n0 = wave * 32;
  const bf16* A = xpb + (size_t)mt * 16 * HD;
  v8f a0 = {}, a1 = {};
#pragma unroll
  for (int k0 = 0; k0 < HD; k0 += 32) {
    v16bf a = load_a_frag(A, HD, k0);
    a0 = wmma_bf16(a, load_b_frag(Wqt, HD, n0, k0), a0);
    a1 = wmma_bf16(a, load_b_frag(Wqt, HD, n0 + 16, k0), a1);
  }
  float v0[8], v1[8];
#pragma unroll
  for (int r = 0; r < 8; ++r) {
    v0[r] = a0[r] + bq[n0 + n];
    v1[r] = a1[r] + bq[n0 + 16 + n];
    float p1 = redsum16(v0[r] + v1[r]);
    float p2 = redsum16(v0[r] * v0[r] + v1[r] * v1[r]);
    if (n == 0) { red1[wave][r + 8 * h] = p1; red2[wave][r + 8 * h] = p2; }
  }
  __syncthreads();
  if (threadIdx.x < 16) {
    float s1 = 0.f, s2 = 0.f;
#pragma unroll
    for (int w = 0; w < 8; ++w) { s1 += red1[w][threadIdx.x]; s2 += red2[w][threadIdx.x]; }
    float mu = s1 / (float)HD, var = s2 / (float)HD - mu * mu;
    smu[threadIdx.x] = mu; srs[threadIdx.x] = rsqrtf(var + EPSLN);
  }
  __syncthreads();
#pragma unroll
  for (int r = 0; r < 8; ++r) {
    int row = r + 8 * h; size_t tok = (size_t)mt * 16 + row;
    float mu = smu[row], rs = srs[row];
    queryb[tok * HD + n0 + n] =
        (bf16)gelu_exact((v0[r] - mu) * rs * gq[n0 + n] + beq[n0 + n]);
    queryb[tok * HD + n0 + 16 + n] =
        (bf16)gelu_exact((v1[r] - mu) * rs * gq[n0 + 16 + n] + beq[n0 + 16 + n]);
  }
}

// ---- Fused memory read: flash-style softmax over S, mem_out, surprise ------
// K/V tiles staged once per workgroup through the CDNA5 async global->LDS path.
__global__ __launch_bounds__(256) void k_attn(
    const bf16* __restrict__ queryb, const bf16* __restrict__ Kb,
    const bf16* __restrict__ Vt, bf16* __restrict__ memb,
    float* __restrict__ surprise) {
  __shared__ __align__(16) bf16 kst[32 * 256];   // K tile: 32 slots x 256
  __shared__ __align__(16) bf16 vst[256 * 32];   // V tile: 256 rows x 32 slots
  __shared__ __align__(32) bf16 pls[8][16 * 32];
  const int tid = threadIdx.x;
  const int wave = tid >> 5, lane = tid & 31;
  const int n = lane & 15, h = lane >> 4;
  const size_t tok0 = (size_t)blockIdx.x * 128 + (size_t)wave * 16;
  v16bf qf[8];
#pragma unroll
  for (int kk = 0; kk < 8; ++kk) qf[kk] = load_a_frag(queryb + tok0 * HD, HD, kk * 32);
  v8f oacc[16];
#pragma unroll
  for (int j = 0; j < 16; ++j) oacc[j] = v8f{};
  float mrow[8], lrow[8];
#pragma unroll
  for (int r = 0; r < 8; ++r) { mrow[r] = -1e30f; lrow[r] = 0.f; }

  for (int s0 = 0; s0 < SS; s0 += 32) {
    __syncthreads();                       // previous tile fully consumed
#pragma unroll
    for (int c4 = 0; c4 < 4; ++c4) {       // 256 lanes x 4 x 16B = 16KB per tile
      int c = tid * 4 + c4;
      async_ld_b128(Kb + (size_t)(s0 + (c >> 5)) * HD + (c & 31) * 8, kst + c * 8);
      async_ld_b128(Vt + (size_t)(c >> 2) * SS + s0 + (c & 3) * 8, vst + c * 8);
    }
    wait_async0();                         // this wave's async copies landed
    __syncthreads();                       // all waves' copies landed

    v8f sa = {}, sb = {};
#pragma unroll
    for (int kk = 0; kk < 8; ++kk) {
      sa = wmma_bf16(qf[kk], load_b_frag(kst, HD, 0, kk * 32), sa);
      sb = wmma_bf16(qf[kk], load_b_frag(kst, HD, 16, kk * 32), sb);
    }
#pragma unroll
    for (int r = 0; r < 8; ++r) {
      float x0 = sa[r] * 0.0625f, x1 = sb[r] * 0.0625f;   // 1/sqrt(256)
      float mnew = fmaxf(mrow[r], redmax16(fmaxf(x0, x1)));
      float corr = __expf(mrow[r] - mnew);
      float p0 = __expf(x0 - mnew), p1 = __expf(x1 - mnew);
      lrow[r] = lrow[r] * corr + redsum16(p0 + p1);
      mrow[r] = mnew;
#pragma unroll
      for (int j = 0; j < 16; ++j) oacc[j][r] *= corr;
      pls[wave][(r + 8 * h) * 32 + n] = (bf16)p0;          // P tile -> LDS
      pls[wave][(r + 8 * h) * 32 + 16 + n] = (bf16)p1;
    }
    v16bf pa = load_a_frag(&pls[wave][0], 32, 0);          // re-layout as A-frag
#pragma unroll
    for (int j = 0; j < 16; ++j)
      oacc[j] = wmma_bf16(pa, load_b_frag(vst, 32, j * 16, 0), oacc[j]);
  }
#pragma unroll
  for (int r = 0; r < 8; ++r) {
    size_t tok = tok0 + r + 8 * h;
    float inv = 1.f / lrow[r];
#pragma unroll
    for (int j = 0; j < 16; ++j)
      memb[tok * HD + j * 16 + n] = (bf16)(oacc[j][r] * inv);
    if (n == 0) surprise[tok] = 1.f - inv;   // 1 - max(attn) = 1 - 1/Z
  }
}

// ---- GEMM3 fused: kv_lr = gelu(LN768(combined @ W_kv + b_kv)), split -------
__global__ __launch_bounds__(256) void k_kv(
    const bf16* __restrict__ xpb, const bf16* __restrict__ memb,
    const bf16* __restrict__ Wkvt, const float* __restrict__ bkv,
    const float* __restrict__ gkv, const float* __restrict__ bekv,
    bf16* __restrict__ keyb, float* __restrict__ valf, float* __restrict__ lr_out) {
  __shared__ float red1[8][16], red2[8][16], smu[16], srs[16], lred[8][16];
  const int mt = blockIdx.x, wave = threadIdx.x >> 5, lane = threadIdx.x & 31;
  const int n = lane & 15, h = lane >> 4, nb = wave * 96;
  v8f acc[6];
#pragma unroll
  for (int j = 0; j < 6; ++j) acc[j] = v8f{};
  for (int k0 = 0; k0 < 512; k0 += 32) {
    const bf16* A = (k0 < 256 ? xpb : memb) + (size_t)mt * 16 * HD;
    v16bf a = load_a_frag(A, HD, k0 & 255);
#pragma unroll
    for (int j = 0; j < 6; ++j)
      acc[j] = wmma_bf16(a, load_b_frag(Wkvt, 512, nb + j * 16, k0), acc[j]);
  }
  float vv[6][8];
#pragma unroll
  for (int r = 0; r < 8; ++r) {
    float p1 = 0.f, p2 = 0.f;
#pragma unroll
    for (int j = 0; j < 6; ++j) {
      float t = acc[j][r] + bkv[nb + j * 16 + n];
      vv[j][r] = t; p1 += t; p2 += t * t;
    }
    p1 = redsum16(p1); p2 = redsum16(p2);
    if (n == 0) { red1[wave][r + 8 * h] = p1; red2[wave][r + 8 * h] = p2; }
  }
  __syncthreads();
  if (threadIdx.x < 16) {
    float s1 = 0.f, s2 = 0.f;
#pragma unroll
    for (int w = 0; w < 8; ++w) { s1 += red1[w][threadIdx.x]; s2 += red2[w][threadIdx.x]; }
    float mu = s1 * (1.f / 768.f), var = s2 * (1.f / 768.f) - mu * mu;
    smu[threadIdx.x] = mu; srs[threadIdx.x] = rsqrtf(var + EPSLN);
  }
  __syncthreads();
  float lrp[8];
#pragma unroll
  for (int r = 0; r < 8; ++r) lrp[r] = 0.f;
#pragma unroll
  for (int r = 0; r < 8; ++r) {
    int row = r + 8 * h; size_t tok = (size_t)mt * 16 + row;
    float mu = smu[row], rs = srs[row];
#pragma unroll
    for (int j = 0; j < 6; ++j) {
      int col = nb + j * 16 + n;
      float t = gelu_exact((vv[j][r] - mu) * rs * gkv[col] + bekv[col]);
      if (col < 256)       keyb[tok * HD + col] = (bf16)t;
      else if (col < 512)  valf[tok * HD + col - 256] = t;
      else                 lrp[r] += 1.f / (1.f + __expf(-t));
    }
  }
#pragma unroll
  for (int r = 0; r < 8; ++r) {
    float s = redsum16(lrp[r]);
    if (n == 0) lred[wave][r + 8 * h] = s;
  }
  __syncthreads();
  if (threadIdx.x < 16) {
    float s = 0.f;
#pragma unroll
    for (int w = 0; w < 8; ++w) s += lred[w][threadIdx.x];
    lr_out[(size_t)mt * 16 + threadIdx.x] = s * (1.f / 256.f);
  }
}

// ---- wsim + argmax over S (never materialize wsim) -------------------------
__global__ __launch_bounds__(256) void k_argmax(
    const bf16* __restrict__ keyb, const bf16* __restrict__ Kb,
    int* __restrict__ idx) {
  const int wave = threadIdx.x >> 5, lane = threadIdx.x & 31;
  const int n = lane & 15, h = lane >> 4;
  const size_t tok0 = (size_t)blockIdx.x * 128 + (size_t)wave * 16;
  v16bf kf[8];
#pragma unroll
  for (int kk = 0; kk < 8; ++kk) kf[kk] = load_a_frag(keyb + tok0 * HD, HD, kk * 32);
  float best[8]; int bidx[8];
#pragma unroll
  for (int r = 0; r < 8; ++r) { best[r] = -1e30f; bidx[r] = 0; }
  for (int s0 = 0; s0 < SS; s0 += 16) {
    v8f s = {};
#pragma unroll
    for (int kk = 0; kk < 8; ++kk)
      s = wmma_bf16(kf[kk], load_b_frag(Kb, HD, s0, kk * 32), s);
#pragma unroll
    for (int r = 0; r < 8; ++r)
      if (s[r] > best[r]) { best[r] = s[r]; bidx[r] = s0 + n; }
  }
#pragma unroll
  for (int r = 0; r < 8; ++r) {
    float bv = best[r]; int bi = bidx[r];
#pragma unroll
    for (int off = 8; off; off >>= 1) {
      float ov = __shfl_xor(bv, off, 16);
      int oi = __shfl_xor(bi, off, 16);
      if (ov > bv || (ov == bv && oi < bi)) { bv = ov; bi = oi; }
    }
    if (n == 0) idx[tok0 + r + 8 * h] = bi;
  }
}

// ---- gated scatter into segment sums ---------------------------------------
__global__ __launch_bounds__(256) void k_scatter(
    const bf16* __restrict__ keyb, const float* __restrict__ valf,
    const float* __restrict__ surprise, const int* __restrict__ idx,
    float* __restrict__ gk, float* __restrict__ gv, float* __restrict__ gcnt) {
  const int t = blockIdx.x, c = threadIdx.x;
  float gate = surprise[t];
  int slot = idx[t];
  atomicAdd(&gk[(size_t)slot * HD + c], gate * (float)keyb[(size_t)t * HD + c]);
  atomicAdd(&gv[(size_t)slot * HD + c], gate * valf[(size_t)t * HD + c]);
  if (c == 0) atomicAdd(&gcnt[slot], gate);
}

__global__ void k_update(const float* __restrict__ Km, const float* __restrict__ Vm,
                         const float* __restrict__ gk, const float* __restrict__ gv,
                         const float* __restrict__ gcnt,
                         float* __restrict__ Ko, float* __restrict__ Vo) {
  int i = blockIdx.x * blockDim.x + threadIdx.x;
  if (i >= SS * HD) return;
  float c = gcnt[i >> 8];
  Ko[i] = Km[i] + WLR * (gk[i] - c * Km[i]);
  Vo[i] = Vm[i] + WLR * (gv[i] - c * Vm[i]);
}

// ---- output: LN(LN(combined @ W_o + b_o) + identity) -----------------------
__global__ __launch_bounds__(256) void k_out(
    const bf16* __restrict__ xpb, const bf16* __restrict__ memb,
    const bf16* __restrict__ Wot, const float* __restrict__ bo,
    const float* __restrict__ go, const float* __restrict__ beo,
    const float* __restrict__ g1, const float* __restrict__ be1,
    const float* __restrict__ xproj, float* __restrict__ outp) {
  __shared__ float red1[8][16], red2[8][16], smu[16], srs[16];
  const int mt = blockIdx.x, wave = threadIdx.x >> 5, lane = threadIdx.x & 31;
  const int n = lane & 15, h = lane >> 4, n0 = wave * 32;
  v8f a0 = {}, a1 = {};
  for (int k0 = 0; k0 < 512; k0 += 32) {
    const bf16* A = (k0 < 256 ? xpb : memb) + (size_t)mt * 16 * HD;
    v16bf a = load_a_frag(A, HD, k0 & 255);
    a0 = wmma_bf16(a, load_b_frag(Wot, 512, n0, k0), a0);
    a1 = wmma_bf16(a, load_b_frag(Wot, 512, n0 + 16, k0), a1);
  }
  float v0[8], v1[8];
#pragma unroll
  for (int r = 0; r < 8; ++r) {
    v0[r] = a0[r] + bo[n0 + n];
    v1[r] = a1[r] + bo[n0 + 16 + n];
    float p1 = redsum16(v0[r] + v1[r]);
    float p2 = redsum16(v0[r] * v0[r] + v1[r] * v1[r]);
    if (n == 0) { red1[wave][r + 8 * h] = p1; red2[wave][r + 8 * h] = p2; }
  }
  __syncthreads();
  if (threadIdx.x < 16) {
    float s1 = 0.f, s2 = 0.f;
#pragma unroll
    for (int w = 0; w < 8; ++w) { s1 += red1[w][threadIdx.x]; s2 += red2[w][threadIdx.x]; }
    float mu = s1 / (float)HD, var = s2 / (float)HD - mu * mu;
    smu[threadIdx.x] = mu; srs[threadIdx.x] = rsqrtf(var + EPSLN);
  }
  __syncthreads();
  float y0[8], y1[8];
#pragma unroll
  for (int r = 0; r < 8; ++r) {
    int row = r + 8 * h; size_t tok = (size_t)mt * 16 + row;
    float mu = smu[row], rs = srs[row];
    y0[r] = (v0[r] - mu) * rs * go[n0 + n] + beo[n0 + n] + xproj[tok * HD + n0 + n];
    y1[r] = (v1[r] - mu) * rs * go[n0 + 16 + n] + beo[n0 + 16 + n] +
            xproj[tok * HD + n0 + 16 + n];
    float p1 = redsum16(y0[r] + y1[r]);
    float p2 = redsum16(y0[r] * y0[r] + y1[r] * y1[r]);
    if (n == 0) { red1[wave][row] = p1; red2[wave][row] = p2; }
  }
  __syncthreads();
  if (threadIdx.x < 16) {
    float s1 = 0.f, s2 = 0.f;
#pragma unroll
    for (int w = 0; w < 8; ++w) { s1 += red1[w][threadIdx.x]; s2 += red2[w][threadIdx.x]; }
    float mu = s1 / (float)HD, var = s2 / (float)HD - mu * mu;
    smu[threadIdx.x] = mu; srs[threadIdx.x] = rsqrtf(var + EPSLN);
  }
  __syncthreads();
#pragma unroll
  for (int r = 0; r < 8; ++r) {
    int row = r + 8 * h; size_t tok = (size_t)mt * 16 + row;
    float mu = smu[row], rs = srs[row];
    outp[tok * HD + n0 + n] = (y0[r] - mu) * rs * g1[n0 + n] + be1[n0 + n];
    outp[tok * HD + n0 + 16 + n] = (y1[r] - mu) * rs * g1[n0 + 16 + n] + be1[n0 + 16 + n];
  }
}

// ---------------------------------------------------------------------------
extern "C" void kernel_launch(void* const* d_in, const int* in_sizes, int n_in,
                              void* d_out, int out_size, void* d_ws, size_t ws_size,
                              hipStream_t stream) {
  (void)in_sizes; (void)n_in; (void)out_size; (void)ws_size;
  const float* x     = (const float*)d_in[0];
  const float* W_in  = (const float*)d_in[1];
  const float* b_in  = (const float*)d_in[2];
  const float* W_q   = (const float*)d_in[3];
  const float* b_q   = (const float*)d_in[4];
  const float* g_q   = (const float*)d_in[5];
  const float* be_q  = (const float*)d_in[6];
  const float* K_mem = (const float*)d_in[7];
  const float* V_mem = (const float*)d_in[8];
  const float* W_kv  = (const float*)d_in[9];
  const float* b_kv  = (const float*)d_in[10];
  const float* g_kv  = (const float*)d_in[11];
  const float* be_kv = (const float*)d_in[12];
  const float* W_o   = (const float*)d_in[13];
  const float* b_o   = (const float*)d_in[14];
  const float* g_o   = (const float*)d_in[15];
  const float* be_o  = (const float*)d_in[16];
  const float* g_1   = (const float*)d_in[17];
  const float* be_1  = (const float*)d_in[18];

  // workspace carve-up
  char* base = (char*)d_ws;
  size_t off = 0;
  auto alloc = [&](size_t bytes) { size_t o = off; off = (off + bytes + 255) & ~(size_t)255; return o; };
  bf16*  xb     = (bf16*)(base + alloc((size_t)BT * HD * 2));
  bf16*  Wint   = (bf16*)(base + alloc((size_t)HD * HD * 2));
  bf16*  Wqt    = (bf16*)(base + alloc((size_t)HD * HD * 2));
  bf16*  Wkvt   = (bf16*)(base + alloc((size_t)768 * 512 * 2));
  bf16*  Wot    = (bf16*)(base + alloc((size_t)HD * 512 * 2));
  bf16*  Kb     = (bf16*)(base + alloc((size_t)SS * HD * 2));
  bf16*  Vt     = (bf16*)(base + alloc((size_t)HD * SS * 2));
  float* xproj  = (float*)(base + alloc((size_t)BT * HD * 4));
  bf16*  xpb    = (bf16*)(base + alloc((size_t)BT * HD * 2));
  bf16*  queryb = (bf16*)(base + alloc((size_t)BT * HD * 2));
  bf16*  memb   = (bf16*)(base + alloc((size_t)BT * HD * 2));
  bf16*  keyb   = (bf16*)(base + alloc((size_t)BT * HD * 2));
  float* valf   = (float*)(base + alloc((size_t)BT * HD * 4));
  float* gk     = (float*)(base + alloc((size_t)SS * HD * 4));
  float* gv     = (float*)(base + alloc((size_t)SS * HD * 4));
  float* gcnt   = (float*)(base + alloc((size_t)SS * 4));
  int*   idx    = (int*)  (base + alloc((size_t)BT * 4));

  // output carve-up: out, K_new, V_new, surprise, lr (flat, in return order)
  float* out_o = (float*)d_out;
  float* Kout  = out_o + (size_t)BT * HD;
  float* Vout  = Kout + (size_t)SS * HD;
  float* sur   = Vout + (size_t)SS * HD;
  float* lr    = sur + BT;

  const int TPB = 256;
  k_cvt<<<(BT * HD + TPB - 1) / TPB, TPB, 0, stream>>>(x, xb, BT * HD);
  k_cvt<<<(SS * HD + TPB - 1) / TPB, TPB, 0, stream>>>(K_mem, Kb, SS * HD);
  k_trans<<<(HD * HD + TPB - 1) / TPB, TPB, 0, stream>>>(W_in, Wint, HD, HD);
  k_trans<<<(HD * HD + TPB - 1) / TPB, TPB, 0, stream>>>(W_q, Wqt, HD, HD);
  k_trans<<<(512 * 768 + TPB - 1) / TPB, TPB, 0, stream>>>(W_kv, Wkvt, 512, 768);
  k_trans<<<(512 * HD + TPB - 1) / TPB, TPB, 0, stream>>>(W_o, Wot, 512, HD);
  k_trans<<<(SS * HD + TPB - 1) / TPB, TPB, 0, stream>>>(V_mem, Vt, SS, HD);

  k_gemm1<<<BT / 16, TPB, 0, stream>>>(xb, Wint, b_in, xproj, xpb);
  k_gemm_q<<<BT / 16, TPB, 0, stream>>>(xpb, Wqt, b_q, g_q, be_q, queryb);
  k_attn<<<BT / 128, TPB, 0, stream>>>(queryb, Kb, Vt, memb, sur);
  k_kv<<<BT / 16, TPB, 0, stream>>>(xpb, memb, Wkvt, b_kv, g_kv, be_kv, keyb, valf, lr);
  k_argmax<<<BT / 128, TPB, 0, stream>>>(keyb, Kb, idx);

  k_zero<<<(SS * HD + TPB - 1) / TPB, TPB, 0, stream>>>(gk, SS * HD);
  k_zero<<<(SS * HD + TPB - 1) / TPB, TPB, 0, stream>>>(gv, SS * HD);
  k_zero<<<(SS + TPB - 1) / TPB, TPB, 0, stream>>>(gcnt, SS);
  k_scatter<<<BT, TPB, 0, stream>>>(keyb, valf, sur, idx, gk, gv, gcnt);
  k_update<<<(SS * HD + TPB - 1) / TPB, TPB, 0, stream>>>(K_mem, V_mem, gk, gv, gcnt,
                                                          Kout, Vout);
  k_out<<<BT / 16, TPB, 0, stream>>>(xpb, memb, Wot, b_o, g_o, be_o, g_1, be_1,
                                     xproj, out_o);
}
